// CompressiveEncoder_21758304321566
// MI455X (gfx1250) — matrix operand order repaired and checked by reading, not compiled, not run on hardware
//
#include <hip/hip_runtime.h>

// ---------------------------------------------------------------------------
// CompressiveEncoder forward for MI455X (gfx1250, wave32, WMMA).
// Everything heavy runs on v_wmma_f32_16x16x32_bf16 (f32 accumulate).
// Weights converted once per call to bf16 (~110MB -> fits 192MB L2).
// GEMM stages tiles with GLOBAL_LOAD_ASYNC_TO_LDS_B128 (ASYNCcnt) and
// double-buffers LDS so the k+1 DMA overlaps the k-tile WMMAs.
// ---------------------------------------------------------------------------

#define N_LAYERS 4
#define D_MODEL  1024
#define HEADS    16
#define D_HEAD   64
#define SEQ      1024
#define FF_DIM   4096
#define MKL      1664          // MAX_KLEN
#define BATCH    4
#define BS       (BATCH * SEQ) // 4096 rows

typedef __bf16          bf16x16 __attribute__((ext_vector_type(16)));
typedef float           f32x8   __attribute__((ext_vector_type(8)));
typedef unsigned short  us8     __attribute__((ext_vector_type(8)));

union FragBF {
    bf16x16        v;
    us8            h[2];
    unsigned short u[16];
};

static __device__ inline f32x8 zero8() {
    f32x8 z = {0.f, 0.f, 0.f, 0.f, 0.f, 0.f, 0.f, 0.f};
    return z;
}

static __device__ inline unsigned short f32_to_bf16(float f) {
    unsigned int u = __float_as_uint(f);
    u += 0x7FFFu + ((u >> 16) & 1u);   // round-to-nearest-even
    return (unsigned short)(u >> 16);
}
static __device__ inline float bf16_to_f32(unsigned short hv) {
    return __uint_as_float(((unsigned int)hv) << 16);
}

static __device__ inline f32x8 wmma_bf16(const FragBF& a, const FragBF& b, f32x8 c) {
    return __builtin_amdgcn_wmma_f32_16x16x32_bf16(false, a.v, false, b.v,
                                                   (short)0, c, false, false);
}

// Flat LDS pointers carry the shared aperture in the high 32 bits; the low 32
// bits are the workgroup-relative byte offset that async-to-LDS VDST expects.
static __device__ inline unsigned lds_off(const void* p) {
    return (unsigned)(unsigned long long)p;
}

// Async DMA global -> LDS, 16B per lane, tracked by ASYNCcnt.
// INST_OFFSET is added to both the global and LDS addresses (ISA 08 §4.4).
static __device__ inline void async_b128(unsigned ldsaddr, unsigned long long gaddr) {
    asm volatile("global_load_async_to_lds_b128 %0, %1, off"
                 :: "v"(ldsaddr), "v"(gaddr) : "memory");
}
static __device__ inline void async_b128_o16(unsigned ldsaddr, unsigned long long gaddr) {
    asm volatile("global_load_async_to_lds_b128 %0, %1, off offset:16"
                 :: "v"(ldsaddr), "v"(gaddr) : "memory");
}

// ---------------------------------------------------------------------------
// fp32 -> bf16 bulk conversion (weights)
// ---------------------------------------------------------------------------
__global__ void cvt_bf16_kernel(const float* __restrict__ src,
                                unsigned short* __restrict__ dst, size_t n) {
    size_t stride = (size_t)gridDim.x * blockDim.x;
    for (size_t i = (size_t)blockIdx.x * blockDim.x + threadIdx.x; i < n; i += stride)
        dst[i] = f32_to_bf16(src[i]);
}

// ---------------------------------------------------------------------------
// Embedding gather: x[b,s,:] = embed[seq[b,s],:]  (fp32 master + bf16 copy)
// ---------------------------------------------------------------------------
__global__ void embed_kernel(const int* __restrict__ seq,
                             const float* __restrict__ emb,
                             float* __restrict__ xf,
                             unsigned short* __restrict__ xb) {
    size_t idx = (size_t)blockIdx.x * blockDim.x + threadIdx.x; // over BS*D
    int d = (int)(idx & (D_MODEL - 1));
    size_t row = idx >> 10;
    int tok = seq[row];
    float v = emb[(size_t)tok * D_MODEL + d];
    xf[idx] = v;
    xb[idx] = f32_to_bf16(v);
}

// ---------------------------------------------------------------------------
// GEMM: C[M,N] = act( A[M,K] * B[N,K]^T + bias[N] )
// A, B bf16; block tile 128x64, 8 waves, K-step 32; WMMA f32 accumulate.
// Double-buffered async global->LDS staging (no VGPR round-trip).
// M % 128 == 0, N % 64 == 0, K % 32 == 0 (always true here).
// ---------------------------------------------------------------------------
__global__ __launch_bounds__(256)
void gemm_bf16_kernel(const unsigned short* __restrict__ A,
                      const unsigned short* __restrict__ Bw,
                      const float* __restrict__ bias,
                      unsigned short* __restrict__ Cb,   // nullable
                      float* __restrict__ Cf,            // nullable
                      int M, int N, int K, int act_gelu) {
    __shared__ __align__(16) unsigned short As[2][128 * 32]; // 2 x 8KB
    __shared__ __align__(16) unsigned short Bs[2][64 * 32];  // 2 x 4KB

    const int t    = threadIdx.x;
    const int wave = t >> 5;
    const int lane = t & 31;
    const int lo   = lane & 15;
    const int hi8  = (lane < 16) ? 0 : 8;
    const int aoff = (lane < 16) ? 0 : 8;   // A-fragment K base
    const int boff = (lane < 16) ? 0 : 8;   // B-fragment K base
    const int m0   = blockIdx.y * 128;
    const int n0   = blockIdx.x * 64;

    f32x8 acc[4];
#pragma unroll
    for (int nc = 0; nc < 4; ++nc) acc[nc] = zero8();

    const int arow = t >> 1, aseg = (t & 1) * 16; // 2 threads/row, 32B per thread
    const int brow = t >> 2, bseg = (t & 3) * 8;  // 4 threads/row, 16B per thread

    // Per thread per tile: 3 async b128 ops (2 for A slice, 1 for B slice).
    auto stage = [&](int kk, int buf) {
        unsigned la = lds_off(&As[buf][arow * 32 + aseg]);
        unsigned long long ga =
            (unsigned long long)(A + (size_t)(m0 + arow) * K + kk + aseg);
        async_b128(la, ga);
        async_b128_o16(la, ga);
        unsigned lb = lds_off(&Bs[buf][brow * 32 + bseg]);
        unsigned long long gb =
            (unsigned long long)(Bw + (size_t)(n0 + brow) * K + kk + bseg);
        async_b128(lb, gb);
    };

    stage(0, 0);
    int cur = 0;
    for (int kk = 0; kk < K; kk += 32) {
        if (kk + 32 < K) {
            stage(kk + 32, cur ^ 1);
            // 3 newest ops may be in flight; per-wave async ops complete in
            // order, so <=3 guarantees the previous tile's 3 ops finished.
            asm volatile("s_wait_asynccnt 3" ::: "memory");
        } else {
            asm volatile("s_wait_asynccnt 0" ::: "memory");
        }
        __syncthreads();

        FragBF af;
        af.h[0] = *(const us8*)&As[cur][(wave * 16 + lo) * 32 + aoff];
        af.h[1] = *(const us8*)&As[cur][(wave * 16 + lo) * 32 + aoff + 16];
#pragma unroll
        for (int nc = 0; nc < 4; ++nc) {
            FragBF bf;
            bf.h[0] = *(const us8*)&Bs[cur][(nc * 16 + lo) * 32 + boff];
            bf.h[1] = *(const us8*)&Bs[cur][(nc * 16 + lo) * 32 + boff + 16];
            acc[nc] = wmma_bf16(af, bf, acc[nc]);
        }
        __syncthreads();   // all reads of buf `cur` done before it is restaged
        cur ^= 1;
    }

#pragma unroll
    for (int nc = 0; nc < 4; ++nc) {
        const int col = n0 + nc * 16 + lo;
        const float bv = bias ? bias[col] : 0.0f;
#pragma unroll
        for (int r = 0; r < 8; ++r) {
            const int row = m0 + wave * 16 + r + hi8;
            float v = acc[nc][r] + bv;
            if (act_gelu) v = 0.5f * v * (1.0f + erff(v * 0.70710678118654752f));
            const size_t addr = (size_t)row * N + col;
            if (Cf) Cf[addr] = v;
            if (Cb) Cb[addr] = f32_to_bf16(v);
        }
    }
}

// ---------------------------------------------------------------------------
// Attention with Music-Transformer skew, flash-style online softmax.
// One wave per (16 query rows, head, batch). All matmuls via WMMA.
//   AC[i,j]  = (q[i]+rwb) . k[j]
//   BD[i,j]  = q[i].re[c] + rbias[c],  c = S-1-i+j,  only for j <= i
//   score    = (AC+BD) * 0.25 ; softmax over full row; O = P @ V
// ---------------------------------------------------------------------------
__global__ __launch_bounds__(32)
void attn_kernel(const unsigned short* __restrict__ qb,   // [B,S,D] bf16
                 const unsigned short* __restrict__ kb,
                 const unsigned short* __restrict__ vb,
                 const unsigned short* __restrict__ reb,  // [H,MKL,DH] bf16 (layer slice)
                 const float* __restrict__ rwb,           // [H,DH]
                 const float* __restrict__ rbias,         // [H,MKL]
                 unsigned short* __restrict__ ob) {       // [B,S,D] bf16
    const int i0   = blockIdx.x * 16;
    const int h    = blockIdx.y;
    const int bb   = blockIdx.z;
    const int lane = threadIdx.x;
    const int lo   = lane & 15;
    const int hi8  = (lane < 16) ? 0 : 8;
    const int kbase = (lane < 16) ? 0 : 8;

    __shared__ __align__(16) float          Bsh[16 * 32]; // skew band spill
    __shared__ __align__(16) unsigned short Psh[16 * 32]; // P bf16 relayout
    __shared__ __align__(16) unsigned short Vsh[64 * 32]; // V transposed [d][j]

    // q fragments for d0 = 0,32 : plain (for BD) and +r_w_bias (for AC)
    FragBF aq[2], abq[2];
    const size_t qrow = ((size_t)bb * SEQ + (i0 + lo)) * D_MODEL + h * D_HEAD;
#pragma unroll
    for (int ks = 0; ks < 2; ++ks) {
        const int d0  = ks * 32;
        const int off = d0 + ((lane < 16) ? 0 : 8);
        aq[ks].h[0] = *(const us8*)(qb + qrow + off);
        aq[ks].h[1] = *(const us8*)(qb + qrow + off + 16);
#pragma unroll
        for (int e = 0; e < 16; ++e) {
            int d = d0 + (e & 7) + ((e >> 3) * 16) + ((lane < 16) ? 0 : 8);
            abq[ks].u[e] = f32_to_bf16(bf16_to_f32(aq[ks].u[e]) + rwb[h * D_HEAD + d]);
        }
    }

    f32x8 o[4];
#pragma unroll
    for (int dc = 0; dc < 4; ++dc) o[dc] = zero8();
    float mrow[8], lrow[8];
#pragma unroll
    for (int r = 0; r < 8; ++r) { mrow[r] = -3.0e38f; lrow[r] = 0.0f; }

    for (int j0 = 0; j0 < SEQ; j0 += 32) {
        // ---- stage V tile transposed: Vsh[d][j], coalesced global reads ----
#pragma unroll
        for (int it = 0; it < 8; ++it) {
            int idx = it * 32 + lane;      // 0..255 -> 32 rows x 8 segs
            int jrow = idx >> 3;
            int seg  = (idx & 7) * 8;
            us8 tmp = *(const us8*)(vb + ((size_t)bb * SEQ + (j0 + jrow)) * D_MODEL
                                    + h * D_HEAD + seg);
#pragma unroll
            for (int e = 0; e < 8; ++e) Vsh[(seg + e) * 32 + jrow] = tmp[e];
        }

        float sc[2][8];
#pragma unroll
        for (int jt = 0; jt < 2; ++jt) {
            const int jsub = j0 + jt * 16;
            // AC via WMMA (K = 64 -> two k-steps)
            f32x8 acc = zero8();
#pragma unroll
            for (int ks = 0; ks < 2; ++ks) {
                const int d0 = ks * 32;
                FragBF bk;
                const size_t krow = ((size_t)bb * SEQ + (jsub + lo)) * D_MODEL
                                  + h * D_HEAD + d0 + kbase;
                bk.h[0] = *(const us8*)(kb + krow);
                bk.h[1] = *(const us8*)(kb + krow + 16);
                acc = wmma_bf16(abq[ks], bk, acc);
            }
            const bool has_bd = (jsub <= i0 + 15);   // tile touches j<=i region
            if (has_bd) {
                const int c0 = SEQ - 16 - i0 + jsub; // band start, >= 0
#pragma unroll
                for (int ct = 0; ct < 2; ++ct) {
                    f32x8 accB = zero8();
                    int c    = c0 + ct * 16 + lo;
                    int crow = (c < SEQ) ? c : (SEQ - 1); // clamp; extras discarded
                    const size_t rr = ((size_t)h * MKL + (MKL - SEQ) + crow) * D_HEAD + kbase;
#pragma unroll
                    for (int ks = 0; ks < 2; ++ks) {
                        const int d0 = ks * 32;
                        FragBF br;
                        br.h[0] = *(const us8*)(reb + rr + d0);
                        br.h[1] = *(const us8*)(reb + rr + d0 + 16);
                        accB = wmma_bf16(aq[ks], br, accB);
                    }
#pragma unroll
                    for (int r = 0; r < 8; ++r)
                        Bsh[(r + hi8) * 32 + ct * 16 + lo] = accB[r];
                }
                __syncthreads();
#pragma unroll
                for (int r = 0; r < 8; ++r) {
                    int iq = i0 + r + hi8;
                    int jk = jsub + lo;
                    float v = acc[r];
                    if (jk <= iq) {
                        int c = SEQ - 1 - iq + jk;   // in [0, S-1]
                        v += Bsh[(r + hi8) * 32 + (c - c0)]
                           + rbias[h * MKL + (MKL - SEQ) + c];
                    }
                    sc[jt][r] = v * 0.25f;           // scale = HEADS^-0.5
                }
                __syncthreads();
            } else {
#pragma unroll
                for (int r = 0; r < 8; ++r) sc[jt][r] = acc[r] * 0.25f;
            }
        }

        // ---- online softmax across the 32 new columns ----
        float al[8], p0[8], p1[8];
#pragma unroll
        for (int r = 0; r < 8; ++r) {
            float mv = fmaxf(sc[0][r], sc[1][r]);
#pragma unroll
            for (int msk = 1; msk < 16; msk <<= 1)
                mv = fmaxf(mv, __shfl_xor(mv, msk, 32));
            float mnew = fmaxf(mrow[r], mv);
            al[r] = __expf(mrow[r] - mnew);
            p0[r] = __expf(sc[0][r] - mnew);
            p1[r] = __expf(sc[1][r] - mnew);
            float sv = p0[r] + p1[r];
#pragma unroll
            for (int msk = 1; msk < 16; msk <<= 1)
                sv += __shfl_xor(sv, msk, 32);
            lrow[r] = lrow[r] * al[r] + sv;
            mrow[r] = mnew;
        }
#pragma unroll
        for (int dc = 0; dc < 4; ++dc)
#pragma unroll
            for (int r = 0; r < 8; ++r) o[dc][r] *= al[r];

        // ---- P (C-layout) -> LDS -> A-fragment, then O += P @ V ----
#pragma unroll
        for (int r = 0; r < 8; ++r) {
            Psh[(r + hi8) * 32 + lo]      = f32_to_bf16(p0[r]);
            Psh[(r + hi8) * 32 + 16 + lo] = f32_to_bf16(p1[r]);
        }
        __syncthreads();
        FragBF pf;
        {
            const int aoff = (lane < 16) ? 0 : 8;
            pf.h[0] = *(const us8*)&Psh[lo * 32 + aoff];
            pf.h[1] = *(const us8*)&Psh[lo * 32 + aoff + 16];
        }
#pragma unroll
        for (int dc = 0; dc < 4; ++dc) {
            FragBF vf;
            const int d = dc * 16 + lo;
            vf.h[0] = *(const us8*)&Vsh[d * 32 + kbase];
            vf.h[1] = *(const us8*)&Vsh[d * 32 + kbase + 16];
            o[dc] = wmma_bf16(pf, vf, o[dc]);
        }
        __syncthreads();  // before Vsh/Psh reuse
    }

    // ---- normalize and write O as [b, i, h*64+d] (== heads-merged layout) ----
#pragma unroll
    for (int dc = 0; dc < 4; ++dc)
#pragma unroll
        for (int r = 0; r < 8; ++r) {
            float outv = o[dc][r] / lrow[r];
            size_t addr = ((size_t)bb * SEQ + (i0 + r + hi8)) * D_MODEL
                        + h * D_HEAD + dc * 16 + lo;
            ob[addr] = f32_to_bf16(outv);
        }
}

// ---------------------------------------------------------------------------
// Residual + LayerNorm: out = LN(x + y) * g + b   (fp32 out + bf16 copy)
// One 256-thread block per row, 4 elems/thread.
// ---------------------------------------------------------------------------
__global__ __launch_bounds__(256)
void resid_ln_kernel(const float* __restrict__ x, const float* __restrict__ y,
                     const float* __restrict__ g, const float* __restrict__ b,
                     float* __restrict__ outf, unsigned short* __restrict__ outb) {
    const int row = blockIdx.x;
    const int t = threadIdx.x;
    float v[4], s = 0.f, s2 = 0.f;
#pragma unroll
    for (int j = 0; j < 4; ++j) {
        int c = t + j * 256;
        float val = x[(size_t)row * D_MODEL + c] + y[(size_t)row * D_MODEL + c];
        v[j] = val; s += val; s2 += val * val;
    }
#pragma unroll
    for (int m = 16; m >= 1; m >>= 1) {
        s += __shfl_xor(s, m, 32);
        s2 += __shfl_xor(s2, m, 32);
    }
    __shared__ float red[16];
    const int wave = t >> 5, lane = t & 31;
    if (lane == 0) { red[wave] = s; red[8 + wave] = s2; }
    __syncthreads();
    s = 0.f; s2 = 0.f;
#pragma unroll
    for (int w = 0; w < 8; ++w) { s += red[w]; s2 += red[8 + w]; }
    const float mu   = s * (1.0f / D_MODEL);
    const float var  = s2 * (1.0f / D_MODEL) - mu * mu;
    const float rstd = rsqrtf(var + 1e-5f);
#pragma unroll
    for (int j = 0; j < 4; ++j) {
        int c = t + j * 256;
        float ov = (v[j] - mu) * rstd * g[c] + b[c];
        outf[(size_t)row * D_MODEL + c] = ov;
        outb[(size_t)row * D_MODEL + c] = f32_to_bf16(ov);
    }
}

// ---------------------------------------------------------------------------
// Host orchestration
// ---------------------------------------------------------------------------
extern "C" void kernel_launch(void* const* d_in, const int* in_sizes, int n_in,
                              void* d_out, int out_size, void* d_ws, size_t ws_size,
                              hipStream_t stream) {
    (void)in_sizes; (void)n_in; (void)out_size; (void)ws_size;

    const int*   seq   = (const int*)  d_in[0];
    // d_in[1] mask: all-true per reference semantics; d_in[2]/[3] cmems/mems unused.
    const float* emb   = (const float*)d_in[4];
    const float* Wq    = (const float*)d_in[5];
    const float* Wk    = (const float*)d_in[6];
    const float* Wv    = (const float*)d_in[7];
    const float* Wo    = (const float*)d_in[8];
    const float* r_emb = (const float*)d_in[9];
    const float* rwb   = (const float*)d_in[10];
    const float* rbias = (const float*)d_in[11];
    const float* ln1g  = (const float*)d_in[12];
    const float* ln1b  = (const float*)d_in[13];
    const float* ln2g  = (const float*)d_in[14];
    const float* ln2b  = (const float*)d_in[15];
    const float* w1    = (const float*)d_in[16];
    const float* b1    = (const float*)d_in[17];
    const float* w2    = (const float*)d_in[18];
    const float* b2    = (const float*)d_in[19];

    // ---- workspace carve (~214 MB total) ----
    char* ws = (char*)d_ws;
    size_t off = 0;
    auto alloc = [&](size_t bytes) -> void* {
        void* p = ws + off;
        off += (bytes + 255) & ~(size_t)255;
        return p;
    };
    float*          xf    = (float*)         alloc((size_t)BS * D_MODEL * 4);
    unsigned short* xb    = (unsigned short*)alloc((size_t)BS * D_MODEL * 2);
    unsigned short* qb    = (unsigned short*)alloc((size_t)BS * D_MODEL * 2);
    unsigned short* kbuf  = (unsigned short*)alloc((size_t)BS * D_MODEL * 2);
    unsigned short* vbuf  = (unsigned short*)alloc((size_t)BS * D_MODEL * 2);
    unsigned short* attnb = (unsigned short*)alloc((size_t)BS * D_MODEL * 2);
    float*          projf = (float*)         alloc((size_t)BS * D_MODEL * 4);
    unsigned short* hb    = (unsigned short*)alloc((size_t)BS * FF_DIM * 2);
    unsigned short* Wqb   = (unsigned short*)alloc((size_t)N_LAYERS * D_MODEL * D_MODEL * 2);
    unsigned short* Wkb   = (unsigned short*)alloc((size_t)N_LAYERS * D_MODEL * D_MODEL * 2);
    unsigned short* Wvb   = (unsigned short*)alloc((size_t)N_LAYERS * D_MODEL * D_MODEL * 2);
    unsigned short* Wob   = (unsigned short*)alloc((size_t)N_LAYERS * D_MODEL * D_MODEL * 2);
    unsigned short* w1b   = (unsigned short*)alloc((size_t)N_LAYERS * FF_DIM * D_MODEL * 2);
    unsigned short* w2b   = (unsigned short*)alloc((size_t)N_LAYERS * D_MODEL * FF_DIM * 2);
    unsigned short* rebb  = (unsigned short*)alloc((size_t)N_LAYERS * HEADS * MKL * D_HEAD * 2);

    // ---- weight conversion to bf16 ----
    const size_t szW  = (size_t)N_LAYERS * D_MODEL * D_MODEL;
    const size_t szW1 = (size_t)N_LAYERS * FF_DIM * D_MODEL;
    const size_t szRe = (size_t)N_LAYERS * HEADS * MKL * D_HEAD;
    cvt_bf16_kernel<<<2048, 256, 0, stream>>>(Wq, Wqb, szW);
    cvt_bf16_kernel<<<2048, 256, 0, stream>>>(Wk, Wkb, szW);
    cvt_bf16_kernel<<<2048, 256, 0, stream>>>(Wv, Wvb, szW);
    cvt_bf16_kernel<<<2048, 256, 0, stream>>>(Wo, Wob, szW);
    cvt_bf16_kernel<<<2048, 256, 0, stream>>>(w1, w1b, szW1);
    cvt_bf16_kernel<<<2048, 256, 0, stream>>>(w2, w2b, szW1);
    cvt_bf16_kernel<<<2048, 256, 0, stream>>>(r_emb, rebb, szRe);

    // ---- embedding ----
    embed_kernel<<<(BS * D_MODEL) / 256, 256, 0, stream>>>(seq, emb, xf, xb);

    const dim3 gD (D_MODEL / 64, BS / 128);   // N=1024 GEMMs
    const dim3 gFF(FF_DIM  / 64, BS / 128);   // N=4096 GEMM
    const dim3 gAt(SEQ / 16, HEADS, BATCH);

    for (int l = 0; l < N_LAYERS; ++l) {
        const size_t wOff  = (size_t)l * D_MODEL * D_MODEL;
        const size_t w1Off = (size_t)l * FF_DIM * D_MODEL;
        const size_t reOff = (size_t)l * HEADS * MKL * D_HEAD;

        // q, k, v projections (bf16 out)
        gemm_bf16_kernel<<<gD, 256, 0, stream>>>(xb, Wqb + wOff, nullptr, qb,   nullptr,
                                                 BS, D_MODEL, D_MODEL, 0);
        gemm_bf16_kernel<<<gD, 256, 0, stream>>>(xb, Wkb + wOff, nullptr, kbuf, nullptr,
                                                 BS, D_MODEL, D_MODEL, 0);
        gemm_bf16_kernel<<<gD, 256, 0, stream>>>(xb, Wvb + wOff, nullptr, vbuf, nullptr,
                                                 BS, D_MODEL, D_MODEL, 0);
        // relative-position attention
        attn_kernel<<<gAt, 32, 0, stream>>>(qb, kbuf, vbuf, rebb + reOff,
                                            rwb + (size_t)l * HEADS * D_HEAD,
                                            rbias + (size_t)l * HEADS * MKL, attnb);
        // output projection (fp32 out for residual)
        gemm_bf16_kernel<<<gD, 256, 0, stream>>>(attnb, Wob + wOff, nullptr, nullptr, projf,
                                                 BS, D_MODEL, D_MODEL, 0);
        resid_ln_kernel<<<BS, 256, 0, stream>>>(xf, projf,
                                                ln1g + (size_t)l * D_MODEL,
                                                ln1b + (size_t)l * D_MODEL, xf, xb);
        // MLP: gelu(x @ w1^T + b1) @ w2^T + b2
        gemm_bf16_kernel<<<gFF, 256, 0, stream>>>(xb, w1b + w1Off,
                                                  b1 + (size_t)l * FF_DIM, hb, nullptr,
                                                  BS, FF_DIM, D_MODEL, 1);
        gemm_bf16_kernel<<<gD, 256, 0, stream>>>(hb, w2b + w1Off,
                                                 b2 + (size_t)l * D_MODEL, nullptr, projf,
                                                 BS, D_MODEL, FF_DIM, 0);
        float* dst = (l == N_LAYERS - 1) ? (float*)d_out : xf;
        resid_ln_kernel<<<BS, 256, 0, stream>>>(xf, projf,
                                                ln2g + (size_t)l * D_MODEL,
                                                ln2b + (size_t)l * D_MODEL, dst, xb);
    }
}
